// LinearAttention_58660663328940
// MI455X (gfx1250) — matrix operand order repaired and checked by reading, not compiled, not run on hardware
//
#include <hip/hip_runtime.h>
#include <hip/hip_bf16.h>

typedef __bf16 bf16;
typedef __attribute__((ext_vector_type(8)))  bf16  bf16x8;
typedef __attribute__((ext_vector_type(16))) bf16  bf16x16;
typedef __attribute__((ext_vector_type(8)))  float floatx8;

#define BATCH  2
#define LSEQ   2048
#define DMODEL 1024
#define HEADS  16
#define DHEAD  64
#define NROWS  (BATCH * LSEQ)   // 4096

// ---------- helpers ----------

static __device__ __forceinline__ bf16 f2bf(float f) {
  unsigned x = __builtin_bit_cast(unsigned, f);
  unsigned r = (x + 0x7FFFu + ((x >> 16) & 1u)) >> 16;   // round-to-nearest-even
  unsigned short u = (unsigned short)r;
  return __builtin_bit_cast(bf16, u);
}

static __device__ __forceinline__ floatx8 wmma_bf16(bf16x16 a, bf16x16 b, floatx8 c) {
  // (neg_a, A, neg_b, B, c_mod, C, reuse_a, reuse_b)
  return __builtin_amdgcn_wmma_f32_16x16x32_bf16(false, a, false, b, (short)0, c,
                                                 false, false);
}

// A fragment: 16x32 (MxK) tile of a row-major matrix, origin `base`, ld elements.
// lanes 0-15: row=lane, K={0..7,16..23}; lanes 16-31: row=lane-16, K={8..15,24..31}
static __device__ __forceinline__ bf16x16 load_a(const bf16* base, int ld, int lane) {
  const int row  = lane & 15;
  const int koff = (lane & 16) ? 8 : 0;
  const bf16* p = base + (size_t)row * ld + koff;
  bf16x8 lo = *(const bf16x8*)(p);
  bf16x8 hi = *(const bf16x8*)(p + 16);
  bf16x16 r;
#pragma unroll
  for (int i = 0; i < 8; ++i) { r[i] = lo[i]; r[8 + i] = hi[i]; }
  return r;
}

// B fragment: 32x16 (KxN) where B[k][n] = M[n][k] of row-major M (op: X @ M^T).
// lanes 0-15: N=lane, K={0..7,16..23}; lanes 16-31: N=lane-16, K={8..15,24..31}.
// Two 16-byte loads so only 16B alignment is required (LDS-safe).
static __device__ __forceinline__ bf16x16 load_bT(const bf16* base, int ld, int lane) {
  const int n    = lane & 15;
  const int koff = (lane & 16) ? 8 : 0;
  const bf16* p = base + (size_t)n * ld + koff;
  bf16x8 lo = *(const bf16x8*)(p);
  bf16x8 hi = *(const bf16x8*)(p + 16);
  bf16x16 r;
#pragma unroll
  for (int i = 0; i < 8; ++i) { r[i] = lo[i]; r[8 + i] = hi[i]; }
  return r;
}

// CDNA5 async global->LDS copy (16B per lane), tracked by ASYNCcnt.
#define ASYNC_B128(ldsoff, gptr)                                              \
  asm volatile("global_load_async_to_lds_b128 %0, %1, off"                    \
               :: "v"(ldsoff),                                                \
                  "v"((unsigned long long)(uintptr_t)(gptr))                  \
               : "memory")

// wait until ASYNCcnt <= n (async loads complete in order -> n=4 waits for the
// older 4 of 8 outstanding ops)
#define WAIT_ASYNC(n) asm volatile("s_wait_asynccnt %0" :: "i"(n) : "memory")

// ---------- fp32 -> bf16 convert ----------

__global__ void cvt_kernel(const float* __restrict__ in, bf16* __restrict__ out, int n) {
  int i = blockIdx.x * 256 + threadIdx.x;
  if (i < n) out[i] = f2bf(in[i]);
}

// ---------- generic  out = X @ W^T  GEMM (bf16 in, f32 accum) ----------
// mode 0: write bf16, head-split  [B,H,L,DH]
// mode 1: write bf16, head-split transposed [B,H,DH,L]   (for V)
// mode 2: write f32 row-major [N, M]                     (final output)
// Explicit ping-pong over K (K must be a multiple of 64; here K==1024):
// loads for one phase fly while the other phase's WMMAs issue, no reg copies.
__global__ __launch_bounds__(256) void gemm_xwT_kernel(
    const bf16* __restrict__ X, const bf16* __restrict__ W,
    void* __restrict__ outp, int K, int M, int mode)
{
  const int tid  = threadIdx.x;
  const int lane = tid & 31;
  const int wave = tid >> 5;
  const int row0 = blockIdx.x * 128 + wave * 16;
  const int col0 = blockIdx.y * 64;

  const bf16* Xrow = X + (size_t)row0 * K;

  floatx8 acc[4] = {};
  bf16x16 a0 = load_a(Xrow, K, lane);
  bf16x16 b0[4];
#pragma unroll
  for (int t = 0; t < 4; ++t)
    b0[t] = load_bT(W + (size_t)(col0 + t * 16) * K, K, lane);
  bf16x16 a1 = a0;
  bf16x16 b1[4] = {b0[0], b0[1], b0[2], b0[3]};

  for (int k0 = 0; k0 < K; k0 += 64) {
    const int kA = k0 + 32;
    if (kA < K) {                        // phase-1 loads under phase-0 WMMAs
      __builtin_prefetch(Xrow + kA + 96, 0, 1);
      a1 = load_a(Xrow + kA, K, lane);
#pragma unroll
      for (int t = 0; t < 4; ++t)
        b1[t] = load_bT(W + (size_t)(col0 + t * 16) * K + kA, K, lane);
    }
#pragma unroll
    for (int t = 0; t < 4; ++t)
      acc[t] = wmma_bf16(a0, b0[t], acc[t]);

    const int kB = k0 + 64;
    if (kB < K) {                        // phase-0 loads under phase-1 WMMAs
      a0 = load_a(Xrow + kB, K, lane);
#pragma unroll
      for (int t = 0; t < 4; ++t)
        b0[t] = load_bT(W + (size_t)(col0 + t * 16) * K + kB, K, lane);
    }
#pragma unroll
    for (int t = 0; t < 4; ++t)
      acc[t] = wmma_bf16(a1, b1[t], acc[t]);
  }

  const int rbase = (lane & 16) ? 8 : 0;
#pragma unroll
  for (int t = 0; t < 4; ++t) {
    const int col = col0 + t * 16 + (lane & 15);
#pragma unroll
    for (int r = 0; r < 8; ++r) {
      const int row = row0 + rbase + r;
      const float v = acc[t][r];
      if (mode == 2) {
        ((float*)outp)[(size_t)row * M + col] = v;
      } else {
        const int b  = row >> 11;             // / LSEQ
        const int l  = row & (LSEQ - 1);
        const int h  = col >> 6;              // / DHEAD
        const int dh = col & (DHEAD - 1);
        if (mode == 0)
          ((bf16*)outp)[(((size_t)(b * HEADS + h)) * LSEQ + l) * DHEAD + dh] = f2bf(v);
        else
          ((bf16*)outp)[(((size_t)(b * HEADS + h)) * DHEAD + dh) * LSEQ + l] = f2bf(v);
      }
    }
  }
}

// ---------- Hadamard feature map:  out = (x@W1^T + b1) * (x@W2^T + b2) ----------
__global__ __launch_bounds__(256) void feat_kernel(
    const bf16* __restrict__ xin,
    const bf16* __restrict__ W1, const float* __restrict__ b1,
    const bf16* __restrict__ W2, const float* __restrict__ b2,
    bf16* __restrict__ outp)
{
  const int tid = threadIdx.x, lane = tid & 31, wave = tid >> 5;
  const size_t row0 = (size_t)blockIdx.x * 128 + (size_t)wave * 16;

  bf16x16 a[2];
  a[0] = load_a(xin + row0 * DHEAD + 0,  DHEAD, lane);
  a[1] = load_a(xin + row0 * DHEAD + 32, DHEAD, lane);

  floatx8 c1[4] = {}, c2[4] = {};
#pragma unroll
  for (int t = 0; t < 4; ++t) {
#pragma unroll
    for (int kk = 0; kk < 2; ++kk) {
      c1[t] = wmma_bf16(a[kk], load_bT(W1 + (size_t)(t * 16) * DHEAD + kk * 32, DHEAD, lane), c1[t]);
      c2[t] = wmma_bf16(a[kk], load_bT(W2 + (size_t)(t * 16) * DHEAD + kk * 32, DHEAD, lane), c2[t]);
    }
  }

  const int rbase = (lane & 16) ? 8 : 0;
#pragma unroll
  for (int t = 0; t < 4; ++t) {
    const int col = t * 16 + (lane & 15);
    const float bb1 = b1[col], bb2 = b2[col];
#pragma unroll
    for (int r = 0; r < 8; ++r) {
      const size_t row = row0 + rbase + r;
      outp[row * DHEAD + col] = f2bf((c1[t][r] + bb1) * (c2[t][r] + bb2));
    }
  }
}

// ---------- causal linear attention (quadratic, tiled) ----------
// qf,kf: [B,H,L,64] bf16 ; vT: [B,H,64,L] bf16 ; y: [B,L,H*64] bf16
// k/v tiles are double-buffered in LDS via async B128 copies: block j+1's
// copies are issued before waiting on block j (in-order ASYNCcnt -> wait<=4),
// hiding the staging latency behind the previous block's WMMAs.
__global__ __launch_bounds__(256) void attn_kernel(
    const bf16* __restrict__ qf, const bf16* __restrict__ kf,
    const bf16* __restrict__ vT, bf16* __restrict__ y)
{
  __shared__ float S [64][65];      // masked scores, f32 (exact z row-sums)
  __shared__ bf16  S2 [64][72];     // bf16 mirror, A-operand for S @ V
  __shared__ bf16  kL [2][64][72];  // k_j tiles  (row = m local, col = dh)
  __shared__ bf16  vL [2][64][72];  // vT_j tiles (row = dh,      col = m local)
  __shared__ float zpart[256];
  __shared__ float zinv[64];

  const int tid  = threadIdx.x;
  const int lane = tid & 31;
  const int wave = tid >> 5;
  const int tr   = wave >> 1;          // 0..3 : 16-row tile
  const int tc0  = (wave & 1) * 2;     // 0 or 2 : first of two 16-col tiles

  const int ib = blockIdx.x;           // row block (0..31)
  const int bh = blockIdx.y;           // 0..31
  const int b  = bh / HEADS, h = bh % HEADS;
  const int n0 = ib * 64;

  const bf16* qh = qf + (size_t)bh * LSEQ * DHEAD;
  const bf16* kh = kf + (size_t)bh * LSEQ * DHEAD;
  const bf16* vh = vT + (size_t)bh * DHEAD * LSEQ;

  // staging assignment: 512 16B segments per tile, 2 per thread per tile
  const int srow  = tid >> 3;          // 0..31
  const int spart = tid & 7;           // 0..7 (16B column chunk)
  const bf16* kg0 = kh + (size_t)srow        * DHEAD + spart * 8;
  const bf16* kg1 = kh + (size_t)(srow + 32) * DHEAD + spart * 8;
  const bf16* vg0 = vh + (size_t)srow        * LSEQ  + spart * 8;
  const bf16* vg1 = vh + (size_t)(srow + 32) * LSEQ  + spart * 8;
  unsigned klds[2][2], vlds[2][2];
#pragma unroll
  for (int p = 0; p < 2; ++p) {
    klds[p][0] = (unsigned)(uintptr_t)&kL[p][srow][spart * 8];
    klds[p][1] = (unsigned)(uintptr_t)&kL[p][srow + 32][spart * 8];
    vlds[p][0] = (unsigned)(uintptr_t)&vL[p][srow][spart * 8];
    vlds[p][1] = (unsigned)(uintptr_t)&vL[p][srow + 32][spart * 8];
  }
  auto stage = [&](int jj) {
    const int p  = jj & 1;
    const int mm = jj * 64;
    ASYNC_B128(klds[p][0], kg0 + (size_t)mm * DHEAD);
    ASYNC_B128(klds[p][1], kg1 + (size_t)mm * DHEAD);
    ASYNC_B128(vlds[p][0], vg0 + mm);
    ASYNC_B128(vlds[p][1], vg1 + mm);
  };

  // q A-fragments for this wave's row tile (constant over the causal loop)
  bf16x16 aq[2];
  aq[0] = load_a(qh + (size_t)(n0 + tr * 16) * DHEAD + 0,  DHEAD, lane);
  aq[1] = load_a(qh + (size_t)(n0 + tr * 16) * DHEAD + 32, DHEAD, lane);

  floatx8 acc[2] = {};                 // y accum: 2 tiles of 16x16 (cols = dh)
  float zp = 0.f;
  const int rbase = (lane & 16) ? 8 : 0;
  const int ncol  = lane & 15;

  stage(0);                            // prologue: stage block 0
  for (int j = 0; j <= ib; ++j) {
    const int cur = j & 1;
    const int m0  = j * 64;

    if (j < ib) { stage(j + 1); WAIT_ASYNC(4); }   // overlap next block's copy
    else        { WAIT_ASYNC(0); }
    __syncthreads();                   // buf[cur] staged by all waves

    // --- S = q k^T from LDS-staged k ---
    floatx8 s[2] = {};
#pragma unroll
    for (int t = 0; t < 2; ++t) {
      const int tc = tc0 + t;
#pragma unroll
      for (int kk = 0; kk < 2; ++kk) {
        bf16x16 bk = load_bT(&kL[cur][tc * 16][kk * 32], 72, lane);
        s[t] = wmma_bf16(aq[kk], bk, s[t]);
      }
    }

    // --- mask + store S (f32 for z, bf16 mirror for WMMA A) ---
#pragma unroll
    for (int t = 0; t < 2; ++t) {
      const int tc = tc0 + t;
#pragma unroll
      for (int r = 0; r < 8; ++r) {
        const int lrow = tr * 16 + rbase + r;
        const int lcol = tc * 16 + ncol;
        float v = s[t][r];
        if (j == ib && (m0 + lcol) > (n0 + lrow)) v = 0.f;   // tril mask
        S [lrow][lcol] = v;
        S2[lrow][lcol] = f2bf(v);
      }
    }
    __syncthreads();                   // S visible

    // --- denominator: z row-sums (== q . cumsum(k) under the mask) ---
    {
      const int rr = tid >> 2, qd = tid & 3;
      float sum = 0.f;
#pragma unroll
      for (int e = 0; e < 16; ++e) sum += S[rr][qd * 16 + e];
      zp += sum;
    }

    // --- y += S @ V_j  (A from LDS bf16 S, B from LDS-staged vT) ---
#pragma unroll
    for (int t = 0; t < 2; ++t) {
      const int d0 = (tc0 + t) * 16;
#pragma unroll
      for (int kk = 0; kk < 2; ++kk) {
        bf16x16 as = load_a(&S2[tr * 16][kk * 32], 72, lane);
        bf16x16 bv = load_bT(&vL[cur][d0][kk * 32], 72, lane);
        acc[t] = wmma_bf16(as, bv, acc[t]);
      }
    }
    __syncthreads();                   // buf[cur] reads done before reuse
  }

  // reduce z across the 4 per-row partials, invert
  zpart[tid] = zp;
  __syncthreads();
  if (tid < 64) {
    float zv = zpart[4 * tid] + zpart[4 * tid + 1] + zpart[4 * tid + 2] + zpart[4 * tid + 3];
    zinv[tid] = 1.f / (zv + 1e-12f);
  }
  __syncthreads();

  // write y * z  to [B, L, H*DH]
#pragma unroll
  for (int t = 0; t < 2; ++t) {
    const int col = (tc0 + t) * 16 + ncol;                   // dh
#pragma unroll
    for (int r = 0; r < 8; ++r) {
      const int lrow = tr * 16 + rbase + r;
      const float v = acc[t][r] * zinv[lrow];
      y[((size_t)b * LSEQ + (n0 + lrow)) * DMODEL + h * DHEAD + col] = f2bf(v);
    }
  }
}

// ---------- launch ----------

extern "C" void kernel_launch(void* const* d_in, const int* in_sizes, int n_in,
                              void* d_out, int out_size, void* d_ws, size_t ws_size,
                              hipStream_t stream)
{
  const float* hidden = (const float*)d_in[0];
  const float* Wq  = (const float*)d_in[1];
  const float* Wk  = (const float*)d_in[2];
  const float* Wv  = (const float*)d_in[3];
  const float* Wo  = (const float*)d_in[4];
  const float* W1q = (const float*)d_in[5];
  const float* b1q = (const float*)d_in[6];
  const float* W2q = (const float*)d_in[7];
  const float* b2q = (const float*)d_in[8];
  const float* W1k = (const float*)d_in[9];
  const float* b1k = (const float*)d_in[10];
  const float* W2k = (const float*)d_in[11];
  const float* b2k = (const float*)d_in[12];

  char* ws = (char*)d_ws;
  size_t off = 0;
  auto alloc = [&](size_t elems) -> bf16* {
    bf16* p = (bf16*)(ws + off);
    off += ((elems * sizeof(bf16) + 255) & ~(size_t)255);
    return p;
  };

  bf16* Xb   = alloc((size_t)NROWS * DMODEL);
  bf16* Wqb  = alloc((size_t)DMODEL * DMODEL);
  bf16* Wkb  = alloc((size_t)DMODEL * DMODEL);
  bf16* Wvb  = alloc((size_t)DMODEL * DMODEL);
  bf16* Wob  = alloc((size_t)DMODEL * DMODEL);
  bf16* W1qb = alloc((size_t)DHEAD * DHEAD);
  bf16* W2qb = alloc((size_t)DHEAD * DHEAD);
  bf16* W1kb = alloc((size_t)DHEAD * DHEAD);
  bf16* W2kb = alloc((size_t)DHEAD * DHEAD);
  bf16* q_bh = alloc((size_t)NROWS * DMODEL);   // [B,H,L,DH]
  bf16* k_bh = alloc((size_t)NROWS * DMODEL);   // [B,H,L,DH]
  bf16* vTb  = alloc((size_t)NROWS * DMODEL);   // [B,H,DH,L]
  bf16* qft  = alloc((size_t)NROWS * DMODEL);   // [B,H,L,DH]
  bf16* kft  = alloc((size_t)NROWS * DMODEL);   // [B,H,L,DH]
  bf16* ybf  = alloc((size_t)NROWS * DMODEL);   // [B,L,H*DH]

  auto cvt = [&](const float* src, bf16* dst, int n) {
    cvt_kernel<<<dim3((n + 255) / 256), dim3(256), 0, stream>>>(src, dst, n);
  };
  cvt(hidden, Xb,   NROWS * DMODEL);
  cvt(Wq,  Wqb,  DMODEL * DMODEL);
  cvt(Wk,  Wkb,  DMODEL * DMODEL);
  cvt(Wv,  Wvb,  DMODEL * DMODEL);
  cvt(Wo,  Wob,  DMODEL * DMODEL);
  cvt(W1q, W1qb, DHEAD * DHEAD);
  cvt(W2q, W2qb, DHEAD * DHEAD);
  cvt(W1k, W1kb, DHEAD * DHEAD);
  cvt(W2k, W2kb, DHEAD * DHEAD);

  const dim3 gemmGrid(NROWS / 128, DMODEL / 64);
  gemm_xwT_kernel<<<gemmGrid, 256, 0, stream>>>(Xb, Wqb, q_bh, DMODEL, DMODEL, 0);
  gemm_xwT_kernel<<<gemmGrid, 256, 0, stream>>>(Xb, Wkb, k_bh, DMODEL, DMODEL, 0);
  gemm_xwT_kernel<<<gemmGrid, 256, 0, stream>>>(Xb, Wvb, vTb,  DMODEL, DMODEL, 1);

  const int featBlocks = (BATCH * HEADS * LSEQ) / 128;   // 512
  feat_kernel<<<dim3(featBlocks), 256, 0, stream>>>(q_bh, W1qb, b1q, W2qb, b2q, qft);
  feat_kernel<<<dim3(featBlocks), 256, 0, stream>>>(k_bh, W1kb, b1k, W2kb, b2k, kft);

  attn_kernel<<<dim3(LSEQ / 64, BATCH * HEADS), 256, 0, stream>>>(qft, kft, vTb, ybf);

  gemm_xwT_kernel<<<gemmGrid, 256, 0, stream>>>(ybf, Wob, d_out, DMODEL, DMODEL, 2);
}
